// Network_47854525612301
// MI455X (gfx1250) — matrix operand order repaired and checked by reading, not compiled, not run on hardware
//
#include <hip/hip_runtime.h>
#include <hip/hip_bf16.h>
#include <math.h>

typedef __attribute__((ext_vector_type(16))) _Float16 v16h;
typedef __attribute__((ext_vector_type(8)))  float    v8f;

#define ROWH 68   // padded LDS row stride (in halves) -> conflict-free column gather

__device__ __forceinline__ float sigmoidf_(float x) { return 1.0f / (1.0f + __expf(-x)); }

// ---------------- kernel 0: pack weights into WMMA fragment order, zero class bits ---------
__global__ __launch_bounds__(256) void setup_kernel(
    const float* __restrict__ W1, const float* __restrict__ b1,
    const float* __restrict__ W2, const float* __restrict__ W3,
    _Float16* __restrict__ frag2, _Float16* __restrict__ frag3,
    float* __restrict__ wbpack, unsigned int* __restrict__ cbits, int C)
{
  int t = blockIdx.x * blockDim.x + threadIdx.x;
  if (t < 2048) {                    // layer-2 B fragments: 64 frags (16 kb x 4 ntile) x 32 lanes
    int frag = t >> 5, lane = t & 31;
    int kb = frag >> 2, nt = frag & 3;
    int n = lane & 15, hi = lane >> 4;
#pragma unroll
    for (int e = 0; e < 16; ++e) {
      int k = kb * 32 + hi * 8 + e + ((e >= 8) ? 8 : 0);   // B-matrix (KxN) lane/element -> K map
      frag2[(frag * 32 + lane) * 16 + e] = (_Float16)W2[(nt * 16 + n) * 512 + k];
    }
  } else if (t < 2048 + 128) {       // layer-3 B fragments: 4 frags (2 kb x 2 ntile) x 32 lanes
    int u = t - 2048;
    int frag = u >> 5, lane = u & 31;
    int kb = frag >> 1, nt = frag & 1;
    int n = lane & 15, hi = lane >> 4;
#pragma unroll
    for (int e = 0; e < 16; ++e) {
      int k = kb * 32 + hi * 8 + e + ((e >= 8) ? 8 : 0);
      frag3[(frag * 32 + lane) * 16 + e] = (_Float16)W3[(nt * 16 + n) * 64 + k];
    }
  } else if (t < 2176 + 1024) {      // W1/b1 packed in A-fragment element order: [kb][hi][w1 x16 | b1 x16]
    int u = t - 2176;
    int kb = u >> 6, rem = u & 63;
    int hi = rem >> 5, w = rem & 31;
    int e = w & 15;
    int j = kb * 32 + hi * 8 + e + ((e >= 8) ? 8 : 0);   // 16-bit A 16x32 K map
    wbpack[u] = (w < 16) ? W1[j] : b1[j];
  } else if (t < 3200 + C) {
    cbits[t - 3200] = 0u;
  }
}

// ---------------- kernel 1: cosine sims (memory-bound streaming pass, 1 wave per row) ------
__global__ __launch_bounds__(256) void sims_kernel(
    const float* __restrict__ enc, const float* __restrict__ query,
    float* __restrict__ sims, int N, int D)
{
  const int lane = threadIdx.x & 31;
  const long gw = (long)((blockIdx.x * blockDim.x + threadIdx.x) >> 5);
  const long nw = (long)((gridDim.x * blockDim.x) >> 5);
  const int C4 = D >> 2;
  const float4* q4 = (const float4*)query;

  float qq = 0.f;
  for (int c = lane; c < C4; c += 32) {
    float4 v = q4[c];
    qq = fmaf(v.x, v.x, fmaf(v.y, v.y, fmaf(v.z, v.z, fmaf(v.w, v.w, qq))));
  }
#pragma unroll
  for (int off = 16; off > 0; off >>= 1) qq += __shfl_xor(qq, off, 32);
  const float qn = fmaxf(sqrtf(qq), 1e-8f);

  for (long row = gw; row < (long)N; row += nw) {
    const float4* e4 = (const float4*)(enc + row * (long)D);
    float dot = 0.f, ee = 0.f;
    for (int c = lane; c < C4; c += 32) {
      float4 e = e4[c]; float4 v = q4[c];
      dot = fmaf(e.x, v.x, fmaf(e.y, v.y, fmaf(e.z, v.z, fmaf(e.w, v.w, dot))));
      ee  = fmaf(e.x, e.x, fmaf(e.y, e.y, fmaf(e.z, e.z, fmaf(e.w, e.w, ee))));
    }
#pragma unroll
    for (int off = 16; off > 0; off >>= 1) {
      dot += __shfl_xor(dot, off, 32);
      ee  += __shfl_xor(ee,  off, 32);
    }
    if (lane == 0) {
      float en = fmaxf(sqrtf(ee), 1e-8f);
      sims[row] = dot / (en * qn);
    }
  }
}

// ---------------- kernel 2: WMMA MLP + atomic segment-max --------------------------------
// One wave handles 64 points (4 groups of 16). Layer2 (512->64) and layer3 (64->32) on
// v_wmma_f32_16x16x32_f16; layer1 is folded into in-register A-fragment construction.
// NOTE: every loop that indexes the acc[][] register arrays is FULLY unrolled so all
// register-array indices are compile-time constants (no scratch demotion).
__global__ __launch_bounds__(256) void mlp_kernel(
    const float* __restrict__ sims, const int* __restrict__ seg,
    const float* __restrict__ wbpack,
    const v16h* __restrict__ frag2, const float* __restrict__ b2,
    const v16h* __restrict__ frag3, const float* __restrict__ b3,
    const float* __restrict__ W4, const float* __restrict__ b4,
    unsigned int* __restrict__ cbits, int N)
{
  __shared__ _Float16 h2buf[8][16 * ROWH];   // per-wave private transpose staging
  const int lane  = threadIdx.x & 31;
  const int wslot = threadIdx.x >> 5;
  const int n  = lane & 15;
  const int hi = lane >> 4;
  const int gw = blockIdx.x * 8 + wslot;
  const int nw = (int)gridDim.x * 8;

  float b2v[4];
#pragma unroll
  for (int nt = 0; nt < 4; ++nt) b2v[nt] = b2[nt * 16 + n];
  const float b3v0 = b3[n], b3v1 = b3[16 + n];
  const float w4a = W4[n], w4b = W4[16 + n];
  const float b4s = b4[0];
  const v8f zero8 = {0.f, 0.f, 0.f, 0.f, 0.f, 0.f, 0.f, 0.f};

  const int iters = (N / 64 + nw - 1) / nw;
  for (int it = 0; it < iters; ++it) {
    const long base = ((long)gw + (long)it * nw) * 64;

    float s[4];
#pragma unroll
    for (int g = 0; g < 4; ++g) {
      long idx = base + g * 16 + n;              // A row m = lane%16
      s[g] = (idx < (long)N) ? sims[idx] : 0.f;
    }

    v8f acc[4][4];
#pragma unroll
    for (int g = 0; g < 4; ++g)
#pragma unroll
      for (int nt = 0; nt < 4; ++nt) acc[g][nt] = zero8;

#pragma unroll 1
    for (int kb = 0; kb < 16; ++kb) {
      const float4* wp = (const float4*)(wbpack + (kb * 2 + hi) * 32);
      float w1v[16], b1v[16];
#pragma unroll
      for (int q = 0; q < 4; ++q) {
        float4 tw = wp[q];
        w1v[q*4+0] = tw.x; w1v[q*4+1] = tw.y; w1v[q*4+2] = tw.z; w1v[q*4+3] = tw.w;
        float4 tb = wp[4 + q];
        b1v[q*4+0] = tb.x; b1v[q*4+1] = tb.y; b1v[q*4+2] = tb.z; b1v[q*4+3] = tb.w;
      }
      // layer-1 computed directly into A fragments (no shuffles needed)
      v16h af[4];
#pragma unroll
      for (int g = 0; g < 4; ++g)
#pragma unroll
        for (int e = 0; e < 16; ++e)
          af[g][e] = (_Float16)fmaxf(0.f, fmaf(w1v[e], s[g], b1v[e]));

#pragma unroll
      for (int nt = 0; nt < 4; ++nt) {
        v16h bf = frag2[(kb * 4 + nt) * 32 + lane];
#pragma unroll
        for (int g = 0; g < 4; ++g)
          acc[g][nt] = __builtin_amdgcn_wmma_f32_16x16x32_f16(
              false, af[g], false, bf, (short)0, acc[g][nt], false, false);
      }
    }

    // -------- epilogue: FULLY unrolled over groups so acc[] indices stay constant --------
#pragma unroll
    for (int g = 0; g < 4; ++g) {
      asm volatile("s_wait_dscnt 0" ::: "memory");   // previous group's LDS reads retired
      // bias + sigmoid, spill h2 (D-layout) to LDS for the layer-3 A-fragment transpose
#pragma unroll
      for (int nt = 0; nt < 4; ++nt)
#pragma unroll
        for (int r = 0; r < 8; ++r) {
          float h2 = sigmoidf_(acc[g][nt][r] + b2v[nt]);
          h2buf[wslot][(r + 8 * hi) * ROWH + nt * 16 + n] = (_Float16)h2;
        }
      asm volatile("s_wait_dscnt 0" ::: "memory");   // stores visible before gathers

      v8f a3acc0 = zero8, a3acc1 = zero8;
#pragma unroll
      for (int kb3 = 0; kb3 < 2; ++kb3) {
        v16h a3;
#pragma unroll
        for (int e = 0; e < 16; ++e) {
          int j = kb3 * 32 + hi * 8 + e + ((e >= 8) ? 8 : 0);
          a3[e] = h2buf[wslot][n * ROWH + j];        // row m = lane%16 gather
        }
        v16h bf0 = frag3[(kb3 * 2 + 0) * 32 + lane];
        v16h bf1 = frag3[(kb3 * 2 + 1) * 32 + lane];
        a3acc0 = __builtin_amdgcn_wmma_f32_16x16x32_f16(false, a3, false, bf0, (short)0, a3acc0, false, false);
        a3acc1 = __builtin_amdgcn_wmma_f32_16x16x32_f16(false, a3, false, bf1, (short)0, a3acc1, false, false);
      }

      // layer 4: per-row dot over 32 outputs, spread over 16 lanes -> xor-butterfly reduce
      float part[8];
#pragma unroll
      for (int r = 0; r < 8; ++r) {
        float x0 = sigmoidf_(a3acc0[r] + b3v0);
        float x1 = sigmoidf_(a3acc1[r] + b3v1);
        part[r] = fmaf(x0, w4a, x1 * w4b);
      }
#pragma unroll
      for (int off = 8; off > 0; off >>= 1)
#pragma unroll
        for (int r = 0; r < 8; ++r) part[r] += __shfl_xor(part[r], off, 32);

      if (n == 0) {                                   // lanes 0 and 16 commit 8 rows each
#pragma unroll
        for (int r = 0; r < 8; ++r) {
          long idx = base + g * 16 + hi * 8 + r;
          if (idx < (long)N) {
            float p = sigmoidf_(part[r] + b4s);       // p > 0 => uint order == float order
            atomicMax(&cbits[seg[idx]], __float_as_uint(p));
          }
        }
      }
    }
  }
}

// ---------------- kernel 3: finalize class_max + first-index argmax -----------------------
__global__ __launch_bounds__(1024) void finalize_kernel(
    const unsigned int* __restrict__ cbits, float* __restrict__ out, int C)
{
  __shared__ float sv[1024];
  __shared__ int   si[1024];
  int t = threadIdx.x;
  float v = -INFINITY;
  if (t < C) {
    unsigned int b = cbits[t];
    v = b ? __uint_as_float(b) : -INFINITY;   // empty segment -> -inf (segment_max identity)
    out[t] = v;
  }
  sv[t] = v; si[t] = t;
  __syncthreads();
  for (int s = 512; s > 0; s >>= 1) {
    if (t < s) {
      if (sv[t + s] > sv[t] || (sv[t + s] == sv[t] && si[t + s] < si[t])) {
        sv[t] = sv[t + s]; si[t] = si[t + s];
      }
    }
    __syncthreads();
  }
  if (t == 0) out[C] = (float)si[0];
}

extern "C" void kernel_launch(void* const* d_in, const int* in_sizes, int n_in,
                              void* d_out, int out_size, void* d_ws, size_t ws_size,
                              hipStream_t stream) {
  const float* query = (const float*)d_in[0];
  const float* enc   = (const float*)d_in[1];
  const int*   seg   = (const int*)d_in[2];
  const float* W1    = (const float*)d_in[4];
  const float* b1    = (const float*)d_in[5];
  const float* W2    = (const float*)d_in[6];
  const float* b2    = (const float*)d_in[7];
  const float* W3    = (const float*)d_in[8];
  const float* b3    = (const float*)d_in[9];
  const float* W4    = (const float*)d_in[10];
  const float* b4    = (const float*)d_in[11];

  const int D = in_sizes[0];      // 1000
  const int N = in_sizes[2];      // 262144
  const int C = out_size - 1;     // 1024 classes + best_class slot

  // workspace layout
  char* ws = (char*)d_ws;
  _Float16*     frag2  = (_Float16*)(ws);                         // 64*512 halves = 64 KB
  _Float16*     frag3  = (_Float16*)(ws + 65536);                 // 4 KB
  float*        wbpack = (float*)   (ws + 65536 + 4096);          // 4 KB
  unsigned int* cbits  = (unsigned int*)(ws + 65536 + 4096 + 4096);
  float*        simsb  = (float*)   (ws + 65536 + 4096 + 4096 + ((C * 4 + 255) & ~255));

  int setup_threads = 3200 + C;
  setup_kernel<<<(setup_threads + 255) / 256, 256, 0, stream>>>(
      W1, b1, W2, W3, frag2, frag3, wbpack, cbits, C);

  sims_kernel<<<1024, 256, 0, stream>>>(enc, query, simsb, N, D);

  mlp_kernel<<<256, 256, 0, stream>>>(
      simsb, seg, wbpack, (const v16h*)frag2, b2, (const v16h*)frag3, b3,
      W4, b4, cbits, N);

  finalize_kernel<<<1, 1024, 0, stream>>>(cbits, (float*)d_out, C);
}
